// StructuralDiffusion3D_31817117728900
// MI455X (gfx1250) — compile-verified, matrix operand order checked by reading, not run on hardware
//
#include <hip/hip_runtime.h>
#include <hip/hip_bf16.h>

typedef __attribute__((ext_vector_type(16))) _Float16 v16h;
typedef __attribute__((ext_vector_type(8)))  _Float16 v8h;
typedef __attribute__((ext_vector_type(8)))  float    v8f;

#define NN 512
#define TT 128
#define HH 192
#define RR 16
#define NHEAD 4
#define HDIM 48

// fast silu: x * v_rcp_f32(1 + e^-x)  (avoids the full-precision divide expansion)
__device__ __forceinline__ float siluf(float x) {
  return x * __builtin_amdgcn_rcpf(1.f + __expf(-x));
}

// ---------------- LayerNorm ----------------
__global__ void ln_kernel(const float* __restrict__ x, const float* __restrict__ g,
                          const float* __restrict__ b, float* __restrict__ y) {
  __shared__ float red[HH];
  __shared__ float mu, rstd;
  int n = blockIdx.x, t = threadIdx.x;
  float v = x[n*HH + t];
  red[t] = v; __syncthreads();
  if (t == 0) { float s = 0.f; for (int i = 0; i < HH; ++i) s += red[i]; mu = s / HH; }
  __syncthreads();
  float d = v - mu; red[t] = d * d; __syncthreads();
  if (t == 0) { float s = 0.f; for (int i = 0; i < HH; ++i) s += red[i]; rstd = rsqrtf(s / HH + 1e-5f); }
  __syncthreads();
  y[n*HH + t] = (v - mu) * rstd * g[t] + b[t];
}

// ---------------- generic small GEMM: Y = act(X@W + bias) + resid ----------------
__global__ void dot_gemm(const float* __restrict__ X, const float* __restrict__ W,
                         const float* __restrict__ bias, const float* __restrict__ resid,
                         float* __restrict__ Y, int M, int K, int Nc, int act) {
  int idx = blockIdx.x * 256 + threadIdx.x;
  if (idx >= M * Nc) return;
  int row = idx / Nc, col = idx % Nc;
  float acc = bias ? bias[col] : 0.f;
  const float* xr = X + (size_t)row * K;
  for (int k = 0; k < K; ++k) acc += xr[k] * W[(size_t)k * Nc + col];
  if (act) acc = siluf(acc);
  if (resid) acc += resid[idx];
  Y[idx] = acc;
}

// node_update layer 1: silu(concat(X0,X1,X2) @ W + b), each X is [M,HH]
__global__ void dot_c3(const float* __restrict__ X0, const float* __restrict__ X1,
                       const float* __restrict__ X2, const float* __restrict__ W,
                       const float* __restrict__ bias, float* __restrict__ Y, int M, int Nc) {
  int idx = blockIdx.x * 256 + threadIdx.x;
  if (idx >= M * Nc) return;
  int row = idx / Nc, col = idx % Nc;
  float acc = bias[col];
  for (int k = 0; k < HH; ++k) acc += X0[row*HH + k] * W[(size_t)k * Nc + col];
  for (int k = 0; k < HH; ++k) acc += X1[row*HH + k] * W[(size_t)(HH + k) * Nc + col];
  for (int k = 0; k < HH; ++k) acc += X2[row*HH + k] * W[(size_t)(2*HH + k) * Nc + col];
  Y[idx] = siluf(acc);
}

// out[n] = bias[n] + sum_h time[h] * W[h*HH + n]   (W already offset to the time-row block)
__global__ void time_fold(const float* __restrict__ W, const float* __restrict__ bias,
                          const float* __restrict__ te, float* __restrict__ out) {
  int n = threadIdx.x;
  float acc = bias[n];
  for (int h = 0; h < HH; ++h) acc += te[h] * W[h*HH + n];
  out[n] = acc;
}

// ---------------- cross attention (4 heads, T=128) ----------------
__global__ void xattn_kernel(const float* __restrict__ q, const float* __restrict__ kk,
                             const float* __restrict__ vv, float* __restrict__ ctx) {
  __shared__ float sc[NHEAD][TT];
  __shared__ float mx[NHEAD], inv[NHEAD];
  int n = blockIdx.x, t = threadIdx.x;
  const float scale = 0.14433756729740643f; // 1/sqrt(48)
  if (t < TT) {
    for (int h = 0; h < NHEAD; ++h) {
      float s = 0.f;
      for (int d = 0; d < HDIM; ++d) s += q[n*HH + h*HDIM + d] * kk[t*HH + h*HDIM + d];
      sc[h][t] = s * scale;
    }
  }
  __syncthreads();
  if (t < NHEAD) {
    float m = -1e30f;
    for (int i = 0; i < TT; ++i) m = fmaxf(m, sc[t][i]);
    float s = 0.f;
    for (int i = 0; i < TT; ++i) s += __expf(sc[t][i] - m);
    mx[t] = m; inv[t] = __builtin_amdgcn_rcpf(s);
  }
  __syncthreads();
  if (t < TT) for (int h = 0; h < NHEAD; ++h) sc[h][t] = __expf(sc[h][t] - mx[h]) * inv[h];
  __syncthreads();
  if (t < HH) {
    int h = t / HDIM;
    float acc = 0.f;
    for (int m = 0; m < TT; ++m) acc += sc[h][m] * vv[m*HH + t];
    ctx[n*HH + t] = acc;
  }
}

// ---------------- pre-fragment weights into WMMA B-operand lane order ----------------
// lane<16 holds k = {0..7,16..23}; lane>=16 holds k = {8..15,24..31}; column n = tile*16 + (lane&15)
__global__ void frag_kernel(const float* __restrict__ eW2, const float* __restrict__ eW1,
                            const float* __restrict__ gW1, _Float16* __restrict__ eW2F,
                            _Float16* __restrict__ e1F, _Float16* __restrict__ g1F) {
  int idx = blockIdx.x * 256 + threadIdx.x;
  if (idx < 12*6*32*16) {
    int h = idx & 15;
    int lane = (idx >> 4) & 31;
    int f = idx >> 9;
    int kc = f % 6, nt = f / 6;
    int n = nt*16 + (lane & 15);
    int kb = (lane < 16) ? (h < 8 ? h : h + 8) : (h < 8 ? h + 8 : h + 16);
    int k = kc*32 + kb;
    eW2F[idx] = (_Float16)eW2[k*HH + n];
  }
  if (idx < 12*32*16) {
    int h = idx & 15;
    int lane = (idx >> 4) & 31;
    int nt = idx >> 9;
    int n = nt*16 + (lane & 15);
    int kb = (lane < 16) ? (h < 8 ? h : h + 8) : (h < 8 ? h + 8 : h + 16);
    e1F[idx] = (kb < RR) ? (_Float16)eW1[(384 + kb)*HH + n] : (_Float16)0.f;
    g1F[idx] = (kb < RR) ? (_Float16)gW1[(384 + kb)*HH + n] : (_Float16)0.f;
  }
}

// ---------------- fused edge kernel: 16 i x 8 j edges per WG, WMMA both MLP layers ----------------
__global__ __launch_bounds__(256) void edge_kernel(
    const float* __restrict__ pos, const float* __restrict__ rbfc,
    const float* __restrict__ Ap, const float* __restrict__ Bp,
    const float* __restrict__ A2p, const float* __restrict__ B2p,
    const _Float16* __restrict__ e1F, const _Float16* __restrict__ g1F,
    const _Float16* __restrict__ eW2F,
    const float* __restrict__ eb2, const float* __restrict__ eW3, const float* __restrict__ eb3p,
    const float* __restrict__ gW2, const float* __restrict__ gb2p,
    float* __restrict__ logits, float* __restrict__ gate) {
  __shared__ _Float16 radf[128*32];   // RBF features, zero padded K=16->32
  __shared__ _Float16 h1s[128*HH];    // silu(layer1) staged for layer-2 GEMM
  __shared__ float bias2[HH], w3[HH], w2g[HH];

  int tid = threadIdx.x;
  int ib = blockIdx.y, jb = blockIdx.x;

  // warm L2/L0 with the fragment weights (global_prefetch_b8)
  __builtin_prefetch(&eW2F[(size_t)tid * 144], 0, 3);
  __builtin_prefetch(&e1F[(size_t)tid * 24], 0, 3);
  __builtin_prefetch(&g1F[(size_t)tid * 24], 0, 3);

  if (tid < HH) { bias2[tid] = eb2[tid]; w3[tid] = eW3[tid]; w2g[tid] = gW2[tid]; }
  if (tid < 128) {
    int gi = ib*16 + (tid >> 3), gj = jb*8 + (tid & 7);
    float dx = pos[gi*3+0] - pos[gj*3+0];
    float dy = pos[gi*3+1] - pos[gj*3+1];
    float dz = pos[gi*3+2] - pos[gj*3+2];
    float dist = sqrtf(fmaxf(dx*dx + dy*dy + dz*dz, 1e-8f));
#pragma unroll
    for (int r = 0; r < RR; ++r) {
      float d = dist - rbfc[r];
      radf[tid*32 + r] = (_Float16)__expf(-1.5f * d * d);
    }
#pragma unroll
    for (int r = RR; r < 32; ++r) radf[tid*32 + r] = (_Float16)0.f;
  }
  __syncthreads();

  int mt   = tid >> 5;          // wave id == M-tile
  int lane = tid & 31;
  int lidx = lane & 15;
  int hh   = lane >> 4;         // half-wave selector for K sub-ranges

  // edges e = mt*16 + hh*8 + r (r=0..7):  i-row is CONSTANT per (wave,half),
  // j-column is exactly r.
  const int gi0 = ib*16 + 2*mt + hh;    // constant source node for this lane's 8 rows
  const int gj0 = jb*8;                 // target nodes gj0..gj0+7

  // A operand (radial) — reused for edge-MLP layer1 and gate layer1
  v16h ar;
  {
    int row = mt*16 + lidx;
    v8h lo = *(const v8h*)&radf[row*32 + hh*8];
    v8h hi = *(const v8h*)&radf[row*32 + 16 + hh*8];
#pragma unroll
    for (int j = 0; j < 8; ++j) { ar[j] = lo[j]; ar[8+j] = hi[j]; }
  }

  float gacc[8];
#pragma unroll
  for (int r = 0; r < 8; ++r) gacc[r] = 0.f;

  for (int nt = 0; nt < 12; ++nt) {
    int n = nt*16 + lidx;
    float a1 = Ap[gi0*HH + n];          // one load instead of 8 (row constant)
    float a2 = A2p[gi0*HH + n];
    const float* bp1 = Bp  + (size_t)gj0*HH + n;   // fixed-stride walk over 8 target rows
    const float* bp2 = B2p + (size_t)gj0*HH + n;
    v8f c1, c2;
#pragma unroll
    for (int r = 0; r < 8; ++r) {
      c1[r] = a1 + bp1[(size_t)r*HH];   // Ap folds eb1 + time term
      c2[r] = a2 + bp2[(size_t)r*HH];   // A2p folds gb1
    }
    v16h b1 = *(const v16h*)&e1F[(nt*32 + lane)*16];
    v16h b2 = *(const v16h*)&g1F[(nt*32 + lane)*16];
    v8f d1 = __builtin_amdgcn_wmma_f32_16x16x32_f16(false, ar, false, b1, (short)0, c1, false, false);
    v8f d2 = __builtin_amdgcn_wmma_f32_16x16x32_f16(false, ar, false, b2, (short)0, c2, false, false);
#pragma unroll
    for (int r = 0; r < 8; ++r) {
      int e = mt*16 + hh*8 + r;
      h1s[e*HH + n] = (_Float16)siluf(d1[r]);      // edge MLP hidden (fp16 for layer-2 GEMM)
      gacc[r] += siluf(d2[r]) * w2g[n];            // gate MLP ends in a dot: no layer-2 GEMM
    }
  }

  // reduce gate partials across the 16 columns held by the half-wave
#pragma unroll
  for (int r = 0; r < 8; ++r) {
    float v = gacc[r];
    v += __shfl_xor(v, 1, 32);
    v += __shfl_xor(v, 2, 32);
    v += __shfl_xor(v, 4, 32);
    v += __shfl_xor(v, 8, 32);
    gacc[r] = v;
  }
  float gb2v = gb2p[0], eb3v = eb3p[0];
  if (lidx == 0) {
#pragma unroll
    for (int r = 0; r < 8; ++r) gate[(size_t)gi0*NN + gj0 + r] = gacc[r] + gb2v;
  }
  __syncthreads();

  // -------- layer 2: [128,192] @ eW2[192,192], then dot with eW3 --------
  float lacc[8];
#pragma unroll
  for (int r = 0; r < 8; ++r) lacc[r] = 0.f;

  for (int nt = 0; nt < 12; ++nt) {
    v8f acc = {0.f,0.f,0.f,0.f,0.f,0.f,0.f,0.f};
#pragma unroll
    for (int kc = 0; kc < 6; ++kc) {
      int row = mt*16 + lidx;
      const _Float16* apd = &h1s[row*HH + kc*32 + hh*8];
      v8h lo = *(const v8h*)apd;
      v8h hi = *(const v8h*)(apd + 16);
      v16h a;
#pragma unroll
      for (int j = 0; j < 8; ++j) { a[j] = lo[j]; a[8+j] = hi[j]; }
      v16h b = *(const v16h*)&eW2F[((nt*6 + kc)*32 + lane)*16];
      acc = __builtin_amdgcn_wmma_f32_16x16x32_f16(false, a, false, b, (short)0, acc, false, false);
    }
    int n = nt*16 + lidx;
#pragma unroll
    for (int r = 0; r < 8; ++r) lacc[r] += siluf(acc[r] + bias2[n]) * w3[n];
  }
#pragma unroll
  for (int r = 0; r < 8; ++r) {
    float v = lacc[r];
    v += __shfl_xor(v, 1, 32);
    v += __shfl_xor(v, 2, 32);
    v += __shfl_xor(v, 4, 32);
    v += __shfl_xor(v, 8, 32);
    lacc[r] = v;
  }
  if (lidx == 0) {
#pragma unroll
    for (int r = 0; r < 8; ++r) {
      int gj = gj0 + r;
      float lg = lacc[r] + eb3v;
      if (gi0 == gj) lg = -1e9f;
      logits[(size_t)gi0*NN + gj] = lg;
    }
  }
}

// ---------------- row softmax over 512 columns ----------------
__global__ void softmax_kernel(const float* __restrict__ L, float* __restrict__ A) {
  __shared__ float red[256];
  int i = blockIdx.x, t = threadIdx.x;
  float a = L[i*NN + t], b = L[i*NN + 256 + t];
  red[t] = fmaxf(a, b); __syncthreads();
  for (int s = 128; s > 0; s >>= 1) { if (t < s) red[t] = fmaxf(red[t], red[t+s]); __syncthreads(); }
  float mx = red[0]; __syncthreads();
  float ea = __expf(a - mx), eb = __expf(b - mx);
  red[t] = ea + eb; __syncthreads();
  for (int s = 128; s > 0; s >>= 1) { if (t < s) red[t] += red[t+s]; __syncthreads(); }
  float inv = __builtin_amdgcn_rcpf(red[0]);
  A[i*NN + t] = ea * inv;
  A[i*NN + 256 + t] = eb * inv;
}

// ---------------- coordinate update: einsum + bias + tanh ----------------
__global__ void coord_kernel(const float* __restrict__ A, const float* __restrict__ G,
                             const float* __restrict__ pos, const float* __restrict__ cb,
                             float* __restrict__ outp) {
  __shared__ float sx[128], sy[128], sz[128];
  int n = blockIdx.x, t = threadIdx.x;
  float px = pos[n*3], py = pos[n*3+1], pz = pos[n*3+2];
  float ax = 0.f, ay = 0.f, az = 0.f;
  for (int m = t; m < NN; m += 128) {
    float w = A[n*NN + m] * G[n*NN + m];
    float dx = px - pos[m*3], dy = py - pos[m*3+1], dz = pz - pos[m*3+2];
    float rinv = rsqrtf(fmaxf(dx*dx + dy*dy + dz*dz, 1e-8f)); // 1/dist
    float s = w * rinv;
    ax += s*dx; ay += s*dy; az += s*dz;
  }
  sx[t] = ax; sy[t] = ay; sz[t] = az; __syncthreads();
  for (int s = 64; s > 0; s >>= 1) {
    if (t < s) { sx[t] += sx[t+s]; sy[t] += sy[t+s]; sz[t] += sz[t+s]; }
    __syncthreads();
  }
  if (t == 0) {
    float cx = sx[0] + 0.1f * cb[n*3+0];
    float cy = sy[0] + 0.1f * cb[n*3+1];
    float cz = sz[0] + 0.1f * cb[n*3+2];
    outp[n*3+0] = px + 0.2f * tanhf(cx);
    outp[n*3+1] = py + 0.2f * tanhf(cy);
    outp[n*3+2] = pz + 0.2f * tanhf(cz);
  }
}

extern "C" void kernel_launch(void* const* d_in, const int* in_sizes, int n_in,
                              void* d_out, int out_size, void* d_ws, size_t ws_size,
                              hipStream_t stream) {
  const float* node_state = (const float*)d_in[0];
  const float* pos        = (const float*)d_in[1];
  const float* latent     = (const float*)d_in[2];
  const float* temb       = (const float*)d_in[3];
  const float* ln_g = (const float*)d_in[4];
  const float* ln_b = (const float*)d_in[5];
  const float* Wq = (const float*)d_in[6];   const float* bq = (const float*)d_in[7];
  const float* Wk = (const float*)d_in[8];   const float* bk = (const float*)d_in[9];
  const float* Wv = (const float*)d_in[10];  const float* bv = (const float*)d_in[11];
  const float* Wo = (const float*)d_in[12];  const float* bo = (const float*)d_in[13];
  const float* eW1 = (const float*)d_in[14]; const float* eb1 = (const float*)d_in[15];
  const float* eW2 = (const float*)d_in[16]; const float* eb2 = (const float*)d_in[17];
  const float* eW3 = (const float*)d_in[18]; const float* eb3 = (const float*)d_in[19];
  const float* nW1 = (const float*)d_in[20]; const float* nb1 = (const float*)d_in[21];
  const float* nW2 = (const float*)d_in[22]; const float* nb2 = (const float*)d_in[23];
  const float* gW1 = (const float*)d_in[24]; const float* gb1 = (const float*)d_in[25];
  const float* gW2 = (const float*)d_in[26]; const float* gb2 = (const float*)d_in[27];
  const float* cW1 = (const float*)d_in[28]; const float* cb1 = (const float*)d_in[29];
  const float* cW2 = (const float*)d_in[30]; const float* cb2 = (const float*)d_in[31];
  const float* rbfc = (const float*)d_in[32];
  (void)in_sizes; (void)n_in; (void)out_size; (void)ws_size;

  size_t off = 0;
  auto alloc = [&](size_t bytes) -> void* {
    off = (off + 255) & ~size_t(255);
    void* p = (char*)d_ws + off;
    off += bytes;
    return p;
  };
  float* normed  = (float*)alloc((size_t)NN*HH*4);
  float* qb      = (float*)alloc((size_t)NN*HH*4);
  float* kb      = (float*)alloc((size_t)TT*HH*4);
  float* vb      = (float*)alloc((size_t)TT*HH*4);
  float* ctx     = (float*)alloc((size_t)NN*HH*4);
  float* xctx    = (float*)alloc((size_t)NN*HH*4);
  float* Apb     = (float*)alloc((size_t)NN*HH*4);
  float* Bpb     = (float*)alloc((size_t)NN*HH*4);
  float* A2pb    = (float*)alloc((size_t)NN*HH*4);
  float* B2pb    = (float*)alloc((size_t)NN*HH*4);
  float* cvec    = (float*)alloc(HH*4);
  float* tvec    = (float*)alloc(HH*4);
  _Float16* eW2F = (_Float16*)alloc((size_t)12*6*32*16*2);
  _Float16* e1F  = (_Float16*)alloc((size_t)12*32*16*2);
  _Float16* g1F  = (_Float16*)alloc((size_t)12*32*16*2);
  float* logits  = (float*)alloc((size_t)NN*NN*4);
  float* gatebuf = (float*)alloc((size_t)NN*NN*4);
  float* attbuf  = (float*)alloc((size_t)NN*NN*4);
  float* nctx    = (float*)alloc((size_t)NN*HH*4);
  float* neigh   = (float*)alloc((size_t)NN*HH*4);
  float* hidden  = (float*)alloc((size_t)NN*2*HH*4);
  float* cbh     = (float*)alloc((size_t)NN*HH*4);
  float* cbias   = (float*)alloc((size_t)NN*3*4);

  float* out_state = (float*)d_out;
  float* out_pos   = (float*)d_out + (size_t)NN*HH;

  // precompute
  ln_kernel<<<NN, HH, 0, stream>>>(node_state, ln_g, ln_b, normed);
  time_fold<<<1, HH, 0, stream>>>(eW1 + 400*HH, eb1, temb, cvec);   // eb1 + time@eW1_time
  time_fold<<<1, HH, 0, stream>>>(cW1 + HH*HH, cb1, temb, tvec);    // cb1 + time@cW1_time
  dot_gemm<<<(NN*HH+255)/256, 256, 0, stream>>>(normed, Wq, bq, nullptr, qb, NN, HH, HH, 0);
  dot_gemm<<<(TT*HH+255)/256, 256, 0, stream>>>(latent, Wk, bk, nullptr, kb, TT, HH, HH, 0);
  dot_gemm<<<(TT*HH+255)/256, 256, 0, stream>>>(latent, Wv, bv, nullptr, vb, TT, HH, HH, 0);
  dot_gemm<<<(NN*HH+255)/256, 256, 0, stream>>>(normed, eW1,        cvec,    nullptr, Apb,  NN, HH, HH, 0);
  dot_gemm<<<(NN*HH+255)/256, 256, 0, stream>>>(normed, eW1+HH*HH,  nullptr, nullptr, Bpb,  NN, HH, HH, 0);
  dot_gemm<<<(NN*HH+255)/256, 256, 0, stream>>>(normed, gW1,        gb1,     nullptr, A2pb, NN, HH, HH, 0);
  dot_gemm<<<(NN*HH+255)/256, 256, 0, stream>>>(normed, gW1+HH*HH,  nullptr, nullptr, B2pb, NN, HH, HH, 0);
  xattn_kernel<<<NN, 256, 0, stream>>>(qb, kb, vb, ctx);
  dot_gemm<<<(NN*HH+255)/256, 256, 0, stream>>>(ctx, Wo, bo, nullptr, xctx, NN, HH, HH, 0);
  frag_kernel<<<(12*6*32*16+255)/256, 256, 0, stream>>>(eW2, eW1, gW1, eW2F, e1F, g1F);

  // fused per-edge MLPs (the ~21 GFLOP WMMA core)
  edge_kernel<<<dim3(64, 32), 256, 0, stream>>>(pos, rbfc, Apb, Bpb, A2pb, B2pb,
                                                e1F, g1F, eW2F, eb2, eW3, eb3, gW2, gb2,
                                                logits, gatebuf);

  // tail
  softmax_kernel<<<NN, 256, 0, stream>>>(logits, attbuf);
  dot_gemm<<<(NN*HH+255)/256, 256, 0, stream>>>(attbuf, xctx,   nullptr, nullptr, nctx,  NN, NN, HH, 0);
  dot_gemm<<<(NN*HH+255)/256, 256, 0, stream>>>(attbuf, normed, nullptr, nullptr, neigh, NN, NN, HH, 0);
  dot_c3<<<(NN*2*HH+255)/256, 256, 0, stream>>>(normed, nctx, neigh, nW1, nb1, hidden, NN, 2*HH);
  dot_gemm<<<(NN*HH+255)/256, 256, 0, stream>>>(hidden, nW2, nb2, node_state, out_state, NN, 2*HH, HH, 0);
  dot_gemm<<<(NN*HH+255)/256, 256, 0, stream>>>(out_state, cW1, tvec, nullptr, cbh, NN, HH, HH, 1);
  dot_gemm<<<(NN*3+255)/256, 256, 0, stream>>>(cbh, cW2, cb2, nullptr, cbias, NN, HH, 3, 0);
  coord_kernel<<<NN, 128, 0, stream>>>(attbuf, gatebuf, pos, cbias, out_pos);
}
